// Attention1D_3719441678428
// MI455X (gfx1250) — matrix operand order repaired and checked by reading, not compiled, not run on hardware
//
#include <hip/hip_runtime.h>

typedef __attribute__((ext_vector_type(16))) __bf16 v16bf;
typedef __attribute__((ext_vector_type(8)))  __bf16 v8bf;
typedef __attribute__((ext_vector_type(8)))  float  v8f;
typedef __attribute__((ext_vector_type(4)))  float  v4f;

// ---------- helpers ----------
__device__ __forceinline__ __bf16 cvt_bf16(float f) {
  return (__bf16)f;   // native fptrunc -> hardware bf16 cvt on gfx1250 (RNE)
}

// 16 contiguous bf16 (16B-aligned) -> v16bf (two 16B loads, avoids 32B align assumption)
__device__ __forceinline__ v16bf load16(const __bf16* p) {
  v8bf lo = *(const v8bf*)p;
  v8bf hb = *(const v8bf*)(p + 8);
  v16bf r;
  #pragma unroll
  for (int e = 0; e < 8; e++) { r[e] = lo[e]; r[e + 8] = hb[e]; }
  return r;
}

// A-fragment gather: elements 0..7 from p, 8..15 from p+8 (contiguous k order)
__device__ __forceinline__ v16bf load_a_frag(const __bf16* p) {
  v8bf lo = *(const v8bf*)p;
  v8bf hb = *(const v8bf*)(p + 8);
  v16bf r;
  #pragma unroll
  for (int e = 0; e < 8; e++) { r[e] = lo[e]; r[e + 8] = hb[e]; }
  return r;
}

// ---------- logmask precompute: lm = log(mask + 1e-30), done ONCE (not per head) ----------
__global__ __launch_bounds__(256) void logmask_kernel(const float* __restrict__ mask,
                                                      float* __restrict__ lm, int n4) {
  int i = blockIdx.x * blockDim.x + threadIdx.x;
  if (i >= n4) return;
  v4f m = *(const v4f*)(mask + 4 * (size_t)i);
  v4f o;
  #pragma unroll
  for (int j = 0; j < 4; j++) o[j] = __logf(m[j] + 1e-30f);
  *(v4f*)(lm + 4 * (size_t)i) = o;
}

// ---------- GEMM: C[M,N] = A[M,K] * W[N,K]^T + bias[N] ----------
// OUT_MODE 0: write bf16 head-split [B,H,S,DK];  OUT_MODE 1: write f32 flat [M,N]
template <typename AT, int OUT_MODE>
__global__ __launch_bounds__(128) void gemm_wmma_kernel(
    const AT* __restrict__ A, const float* __restrict__ W,
    const float* __restrict__ bias, void* __restrict__ outp,
    int M, int N, int Kd) {
  constexpr int BM = 64, BN = 64, BK = 32, LDP = 40;  // pitch 40 bf16 = 80B (16B mult)
  __shared__ alignas(16) __bf16 As[BM][LDP];
  __shared__ alignas(16) __bf16 Bs[BN][LDP];

  const int tid  = threadIdx.x;
  const int lane = tid & 31;
  const int hi   = (lane >> 4) & 1;
  const int ln   = lane & 15;
  const int w    = tid >> 5;          // wave 0..3
  const int wm   = (w >> 1) * 32;     // wave m-offset in block tile
  const int wn   = (w & 1) * 32;      // wave n-offset
  const int m0   = blockIdx.y * BM;
  const int n0   = blockIdx.x * BN;

  v8f acc[2][2];
  #pragma unroll
  for (int i = 0; i < 2; i++)
    #pragma unroll
    for (int j = 0; j < 2; j++)
      #pragma unroll
      for (int r = 0; r < 8; r++) acc[i][j][r] = 0.0f;

  const int srow = tid >> 1;          // 0..63
  const int scol = (tid & 1) * 16;    // 0 or 16

  for (int k0 = 0; k0 < Kd; k0 += BK) {
    __syncthreads();
    // stage A tile (convert f32->bf16 if needed)
    if constexpr (sizeof(AT) == 4) {
      const float* src = (const float*)A + (size_t)(m0 + srow) * Kd + k0 + scol;
      if (k0 + 2 * BK <= Kd - BK) __builtin_prefetch(src + 2 * BK, 0, 1);
      #pragma unroll
      for (int i = 0; i < 16; i += 4) {
        v4f f = *(const v4f*)(src + i);
        #pragma unroll
        for (int j = 0; j < 4; j++) As[srow][scol + i + j] = cvt_bf16(f[j]);
      }
    } else {
      const __bf16* src = (const __bf16*)A + (size_t)(m0 + srow) * Kd + k0 + scol;
      if (k0 + 2 * BK <= Kd - BK) __builtin_prefetch(src + 2 * BK, 0, 1);
      *(v8bf*)&As[srow][scol]     = *(const v8bf*)src;
      *(v8bf*)&As[srow][scol + 8] = *(const v8bf*)(src + 8);
    }
    // stage W tile (f32 -> bf16)
    {
      const float* src = W + (size_t)(n0 + srow) * Kd + k0 + scol;
      if (k0 + 2 * BK <= Kd - BK) __builtin_prefetch(src + 2 * BK, 0, 1);
      #pragma unroll
      for (int i = 0; i < 16; i += 4) {
        v4f f = *(const v4f*)(src + i);
        #pragma unroll
        for (int j = 0; j < 4; j++) Bs[srow][scol + i + j] = cvt_bf16(f[j]);
      }
    }
    __syncthreads();

    v16bf afrag[2], bfrag[2];
    #pragma unroll
    for (int mt = 0; mt < 2; mt++)
      afrag[mt] = load_a_frag(&As[wm + mt * 16 + ln][16 * hi]);
    #pragma unroll
    for (int nt = 0; nt < 2; nt++)
      bfrag[nt] = load16(&Bs[wn + nt * 16 + ln][16 * hi]);

    #pragma unroll
    for (int mt = 0; mt < 2; mt++)
      #pragma unroll
      for (int nt = 0; nt < 2; nt++)
        acc[mt][nt] = __builtin_amdgcn_wmma_f32_16x16x32_bf16(
            false, afrag[mt], false, bfrag[nt], (short)0, acc[mt][nt], false, false);
  }

  // epilogue
  #pragma unroll
  for (int mt = 0; mt < 2; mt++)
    #pragma unroll
    for (int nt = 0; nt < 2; nt++)
      #pragma unroll
      for (int r = 0; r < 8; r++) {
        int row = m0 + wm + mt * 16 + r + 8 * hi;
        int col = n0 + wn + nt * 16 + ln;
        float vres = acc[mt][nt][r] + bias[col];
        if constexpr (OUT_MODE == 0) {
          int b = row >> 11, s = row & 2047;        // S = 2048
          int hh = col >> 6, dk = col & 63;         // DK = 64
          ((__bf16*)outp)[(((size_t)b * 16 + hh) * 2048 + s) * 64 + dk] = cvt_bf16(vres);
        } else {
          ((float*)outp)[(size_t)row * N + col] = vres;
        }
      }
}

// ---------- Flash attention over Q/K/V [B,H,S,64] bf16, lmask [B,S,S] f32 ----------
__global__ __launch_bounds__(128) void attn_kernel(
    const __bf16* __restrict__ Q, const __bf16* __restrict__ Kh,
    const __bf16* __restrict__ V, const float* __restrict__ lmask,
    __bf16* __restrict__ X) {
  constexpr int S = 2048, DK = 64;
  __shared__ alignas(16) __bf16 Kt[32][72];   // [key][dk], pitch 72 (144B)
  __shared__ alignas(16) __bf16 Vt[64][40];   // transposed: [dk][key], pitch 40 (80B)
  __shared__ alignas(16) __bf16 Ps[4][16][40];// per-wave P re-layout buffer

  const int tid  = threadIdx.x;
  const int lane = tid & 31;
  const int hi   = (lane >> 4) & 1;
  const int ln   = lane & 15;
  const int w    = tid >> 5;
  const int bh   = blockIdx.y;         // b*16 + h
  const int b    = bh >> 4;
  const int h    = bh & 15;
  const int q0w  = blockIdx.x * 64 + w * 16;  // this wave's 16 query rows

  // Q fragments (16x64, two K=32 chunks), loaded once
  v16bf qf[2];
  {
    const __bf16* qp = Q + ((size_t)bh * S + q0w + ln) * DK;
    #pragma unroll
    for (int c = 0; c < 2; c++) qf[c] = load_a_frag(qp + c * 32 + 16 * hi);
  }

  float m[8], l[8];
  v8f O[4];
  #pragma unroll
  for (int r = 0; r < 8; r++) { m[r] = -1e30f; l[r] = 0.0f; }
  #pragma unroll
  for (int t = 0; t < 4; t++)
    #pragma unroll
    for (int r = 0; r < 8; r++) O[t][r] = 0.0f;

  for (int kt = 0; kt < S / 32; kt++) {
    const int k0 = kt * 32;
    __syncthreads();
    {   // stage K tile [32 keys][64 dk] via CDNA5 async global->LDS copy (ASYNCcnt)
      int key = tid >> 2, dp = (tid & 3) * 16;
      const __bf16* kp = Kh + ((size_t)bh * S + k0 + key) * DK + dp;
      unsigned l0 = (unsigned)(size_t)(void*)&Kt[key][dp];
      unsigned l1 = (unsigned)(size_t)(void*)&Kt[key][dp + 8];
      asm volatile("global_load_async_to_lds_b128 %0, %1, off"
                   :: "v"(l0), "v"(kp) : "memory");
      asm volatile("global_load_async_to_lds_b128 %0, %1, off"
                   :: "v"(l1), "v"(kp + 8) : "memory");
    }
    {   // stage V tile transposed [64 dk][32 keys] (needs transpose -> manual path)
      int key = tid & 31, d0 = (tid >> 5) * 16;
      const __bf16* vp = V + ((size_t)bh * S + k0 + key) * DK + d0;
      v8bf a = *(const v8bf*)vp;
      v8bf c = *(const v8bf*)(vp + 8);
      #pragma unroll
      for (int i = 0; i < 8; i++) { Vt[d0 + i][key] = a[i]; Vt[d0 + 8 + i][key] = c[i]; }
    }
    asm volatile("s_wait_asynccnt 0" ::: "memory");
    __syncthreads();

    // scores: 16 queries x 32 keys = two 16x16 C tiles, K-dim 64 = 2 chunks
    v8f sc[2];
    #pragma unroll
    for (int nt = 0; nt < 2; nt++)
      #pragma unroll
      for (int r = 0; r < 8; r++) sc[nt][r] = 0.0f;
    #pragma unroll
    for (int nt = 0; nt < 2; nt++)
      #pragma unroll
      for (int c = 0; c < 2; c++) {
        v16bf kb = load16(&Kt[nt * 16 + ln][c * 32 + 16 * hi]);
        sc[nt] = __builtin_amdgcn_wmma_f32_16x16x32_bf16(
            false, qf[c], false, kb, (short)0, sc[nt], false, false);
      }

    // scale + precomputed log-mask (L2-resident, reused by all 16 heads)
    float p[2][8];
    #pragma unroll
    for (int nt = 0; nt < 2; nt++)
      #pragma unroll
      for (int r = 0; r < 8; r++) {
        int qr  = q0w + r + 8 * hi;
        int key = k0 + nt * 16 + ln;
        p[nt][r] = sc[nt][r] * 0.125f + lmask[((size_t)b * S + qr) * S + key];
      }

    // online softmax per row (rows live in 16-lane halves; reduce within width 16)
    #pragma unroll
    for (int r = 0; r < 8; r++) {
      float lm = fmaxf(p[0][r], p[1][r]);
      #pragma unroll
      for (int off = 8; off; off >>= 1) lm = fmaxf(lm, __shfl_xor(lm, off, 16));
      float mn    = fmaxf(m[r], lm);
      float alpha = __expf(m[r] - mn);
      float e0 = __expf(p[0][r] - mn);
      float e1 = __expf(p[1][r] - mn);
      float rs = e0 + e1;
      #pragma unroll
      for (int off = 8; off; off >>= 1) rs += __shfl_xor(rs, off, 16);
      l[r] = l[r] * alpha + rs;
      m[r] = mn;
      #pragma unroll
      for (int t = 0; t < 4; t++) O[t][r] *= alpha;
      p[0][r] = e0; p[1][r] = e1;
    }

    // P: C-layout -> LDS -> A-fragment layout (wave-private region, DS in-order)
    #pragma unroll
    for (int nt = 0; nt < 2; nt++)
      #pragma unroll
      for (int r = 0; r < 8; r++)
        Ps[w][r + 8 * hi][nt * 16 + ln] = cvt_bf16(p[nt][r]);
    v16bf pf = load_a_frag(&Ps[w][ln][16 * hi]);

    // O += P (16x32) x V (32x64): 4 dk tiles
    #pragma unroll
    for (int t = 0; t < 4; t++) {
      v16bf vf = load16(&Vt[t * 16 + ln][16 * hi]);
      O[t] = __builtin_amdgcn_wmma_f32_16x16x32_bf16(
          false, pf, false, vf, (short)0, O[t], false, false);
    }
  }

  // normalize and write X [B,S,D] bf16 (flat, heads merged)
  #pragma unroll
  for (int t = 0; t < 4; t++)
    #pragma unroll
    for (int r = 0; r < 8; r++) {
      int qr  = q0w + r + 8 * hi;
      int col = h * 64 + t * 16 + ln;
      X[((size_t)b * S + qr) * 1024 + col] = cvt_bf16(O[t][r] / l[r]);
    }
}

// ---------- launcher ----------
extern "C" void kernel_launch(void* const* d_in, const int* in_sizes, int n_in,
                              void* d_out, int out_size, void* d_ws, size_t ws_size,
                              hipStream_t stream) {
  (void)in_sizes; (void)n_in; (void)out_size; (void)ws_size;
  const float* query = (const float*)d_in[0];
  const float* key   = (const float*)d_in[1];
  const float* value = (const float*)d_in[2];
  const float* smask = (const float*)d_in[3];
  const float* Wq = (const float*)d_in[4];
  const float* bq = (const float*)d_in[5];
  const float* Wk = (const float*)d_in[6];
  const float* bk = (const float*)d_in[7];
  const float* Wv = (const float*)d_in[8];
  const float* bv = (const float*)d_in[9];
  const float* Wo = (const float*)d_in[10];
  const float* bo = (const float*)d_in[11];
  float* out = (float*)d_out;

  const size_t BHSDK = (size_t)2 * 16 * 2048 * 64;   // 4.19M elems each (bf16)
  __bf16* Qh = (__bf16*)d_ws;
  __bf16* Kh = Qh + BHSDK;
  __bf16* Vh = Kh + BHSDK;
  __bf16* X  = Vh + BHSDK;                           // [B,S,D] bf16
  float*  LM = (float*)(X + BHSDK);                  // [B,S,S] f32 logmask (33.5 MB)

  const int n4 = (2 * 2048 * 2048) / 4;
  logmask_kernel<<<(n4 + 255) / 256, 256, 0, stream>>>(smask, LM, n4);

  dim3 gg(1024 / 64, 4096 / 64);  // (N tiles, M tiles)
  gemm_wmma_kernel<float, 0><<<gg, 128, 0, stream>>>(query, Wq, bq, Qh, 4096, 1024, 1024);
  gemm_wmma_kernel<float, 0><<<gg, 128, 0, stream>>>(key,   Wk, bk, Kh, 4096, 1024, 1024);
  gemm_wmma_kernel<float, 0><<<gg, 128, 0, stream>>>(value, Wv, bv, Vh, 4096, 1024, 1024);
  attn_kernel<<<dim3(2048 / 64, 32), 128, 0, stream>>>(Qh, Kh, Vh, LM, X);
  gemm_wmma_kernel<__bf16, 1><<<gg, 128, 0, stream>>>(X, Wo, bo, out, 4096, 1024, 1024);
}
